// LSTM_NET_36971078484641
// MI455X (gfx1250) — compile-verified, matrix operand order checked by reading, not run on hardware
//
#include <hip/hip_runtime.h>
#include <hip/hip_bf16.h>

// ---------------------------------------------------------------------------
// Problem constants (match reference)
// ---------------------------------------------------------------------------
#define BATCH 4096
#define TSTEPS 128
#define HID 96
#define INF 24
#define G4H 384            // 4*HID
#define BT 64              // batch tile per workgroup
#define NTHREADS 256       // 8 waves (wave32)

typedef __attribute__((__ext_vector_type__(16))) __bf16 v16bf;
typedef __attribute__((__ext_vector_type__(8)))  __bf16 bf16x8;
typedef __attribute__((__ext_vector_type__(8)))  float  v8f;

// ---------------------------------------------------------------------------
// Device helpers
// ---------------------------------------------------------------------------
__device__ __forceinline__ float bf2f(unsigned short u) {
    return __uint_as_float(((unsigned int)u) << 16);
}
__device__ __forceinline__ unsigned short f2bf(float f) {
    __bf16 h = (__bf16)f;
    unsigned short u;
    __builtin_memcpy(&u, &h, 2);
    return u;
}
__device__ __forceinline__ float fast_sigmoid(float x) {
    return 1.0f / (1.0f + __expf(-x));
}
__device__ __forceinline__ float fast_tanh(float x) {
    float ax = __builtin_fabsf(x);
    float e  = __expf(-2.0f * ax);
    float t  = (1.0f - e) / (1.0f + e);
    return __builtin_copysignf(t, x);
}

// Load a 16x32 bf16 A-fragment (or K x N B-fragment of W^T, which by
// B[k][n] = W[n][k] uses the identical per-lane gather from row-major W):
// lane L holds row r0+(L&15); its 16 values are two contiguous 8-elem
// (16-byte) chunks at k0+(L>=16?8:0) and k0+16+(L>=16?8:0)  -> 2x ds_load_b128
__device__ __forceinline__ v16bf ld_frag(const __bf16* m, int r0, int k0, int ld) {
    const int lane = threadIdx.x & 31;
    const int r    = r0 + (lane & 15);
    const int kb   = k0 + ((lane & 16) ? 8 : 0);
    const __bf16* p = m + r * ld + kb;
    bf16x8 lo = *reinterpret_cast<const bf16x8*>(p);
    bf16x8 hi = *reinterpret_cast<const bf16x8*>(p + 16);
    v16bf out;
#pragma unroll
    for (int i = 0; i < 8; ++i) { out[i] = lo[i]; out[i + 8] = hi[i]; }
    return out;
}

// ---------------------------------------------------------------------------
// Prep kernels: f32 -> bf16 conversions / padding / bias fold
// ---------------------------------------------------------------------------
__global__ void k_cvt_flat(const float* __restrict__ s, unsigned short* __restrict__ d, int n) {
    int i = blockIdx.x * blockDim.x + threadIdx.x;
    if (i < n) d[i] = f2bf(s[i]);
}

// W_ih0 [384,24] f32 -> [384,32] bf16 (zero-padded K)
__global__ void k_cvt_wih0(const float* __restrict__ s, unsigned short* __restrict__ d) {
    int i = blockIdx.x * blockDim.x + threadIdx.x;
    if (i < G4H * 32) {
        int k = i & 31, n = i >> 5;
        d[i] = (k < INF) ? f2bf(s[n * INF + k]) : (unsigned short)0;
    }
}

__global__ void k_bias_add(const float* __restrict__ a, const float* __restrict__ b,
                           float* __restrict__ d, int n) {
    int i = blockIdx.x * blockDim.x + threadIdx.x;
    if (i < n) d[i] = a[i] + b[i];
}

// x [B,T,24] f32 -> xpad [T,B,32] bf16 (transpose + zero-pad K)
__global__ void k_cvt_x(const float* __restrict__ x, unsigned short* __restrict__ d) {
    int i = blockIdx.x * blockDim.x + threadIdx.x;   // over T*B*32
    int t = i >> 17;                                 // / (4096*32)
    int r = i & ((BATCH * 32) - 1);
    int b = r >> 5;
    int k = r & 31;
    d[i] = (k < INF) ? f2bf(x[((size_t)b * TSTEPS + t) * INF + k]) : (unsigned short)0;
}

// ---------------------------------------------------------------------------
// Fused per-layer LSTM kernel.
//   Each WG: 64 batch rows, all 128 timesteps. Weights + h in LDS, c in VGPRs.
//   gates[64,384] = x_t[64,KIN] @ W_ih^T + h[64,96] @ W_hh^T + bias
//   Wave task (mt,jt) computes all four gate n-tiles (i/f/g/o) for the same
//   hidden columns -> cell update entirely in registers.
//   The x-tile for step t+1 is loaded into VGPRs while step t computes
//   (software pipeline): the L2 round-trip hides under the WMMA phase.
// ---------------------------------------------------------------------------
template <int KIN>
__global__ __launch_bounds__(NTHREADS) void k_lstm_layer(
    const unsigned short* __restrict__ ysin_u,   // [T][B][KIN] bf16
    unsigned short* __restrict__ ysout_u,        // [T][B][96]  bf16 (may alias ysin)
    const unsigned short* __restrict__ wih_u,    // [384][KIN]  bf16
    const unsigned short* __restrict__ whh_u,    // [384][96]   bf16
    const float* __restrict__ bias,              // [384] f32 (b_ih+b_hh)
    unsigned short* __restrict__ hn_out_u)       // [B][96] bf16 or nullptr
{
    extern __shared__ char smem_raw[];
    __bf16* s_wih = reinterpret_cast<__bf16*>(smem_raw);        // 384*KIN
    __bf16* s_whh = s_wih + G4H * KIN;                          // 384*96
    __bf16* s_h   = s_whh + G4H * HID;                          // 64*96
    __bf16* s_x   = s_h + BT * HID;                             // 64*KIN
    float*  s_b   = reinterpret_cast<float*>(s_x + BT * KIN);   // 384

    const int tid  = threadIdx.x;
    const int lane = tid & 31;
    const int wv   = tid >> 5;
    const int b0   = blockIdx.x * BT;

    // --- Stage weights & bias into LDS (coalesced dword copies) ---
    {
        const unsigned int* g = reinterpret_cast<const unsigned int*>(wih_u);
        unsigned int* s = reinterpret_cast<unsigned int*>(s_wih);
        for (int i = tid; i < G4H * KIN / 2; i += NTHREADS) s[i] = g[i];
        g = reinterpret_cast<const unsigned int*>(whh_u);
        s = reinterpret_cast<unsigned int*>(s_whh);
        for (int i = tid; i < G4H * HID / 2; i += NTHREADS) s[i] = g[i];
        for (int i = tid; i < G4H; i += NTHREADS) s_b[i] = bias[i];
        // zero-init h
        s = reinterpret_cast<unsigned int*>(s_h);
        for (int i = tid; i < BT * HID / 2; i += NTHREADS) s[i] = 0u;
    }

    // 24 tasks = 4 m-tiles x 6 j-tiles; 3 tasks per wave
    int mt_[3], jt_[3];
#pragma unroll
    for (int s = 0; s < 3; ++s) {
        int task = wv + 8 * s;
        mt_[s] = task & 3;
        jt_[s] = task >> 2;
    }

    float creg[3][8];
    float hreg[3][8];
#pragma unroll
    for (int s = 0; s < 3; ++s)
#pragma unroll
        for (int v = 0; v < 8; ++v) creg[s][v] = 0.0f;

    const unsigned int* ysin32 = reinterpret_cast<const unsigned int*>(ysin_u);
    unsigned int* ysout32 = reinterpret_cast<unsigned int*>(ysout_u);

    // ---- software pipeline: x-tile for step t held in registers ----
    constexpr int NX = BT * KIN / 2 / NTHREADS;   // dwords per thread per tile
    unsigned int xr[NX];
    {
        const unsigned int* src = ysin32 + (size_t)b0 * (KIN / 2);   // t = 0
#pragma unroll
        for (int i = 0; i < NX; ++i) xr[i] = src[tid + i * NTHREADS];
    }
    __syncthreads();   // weights/h staged

    for (int t = 0; t < TSTEPS; ++t) {
        // ---- commit x_t tile (registers -> LDS) ----
        {
            unsigned int* dst = reinterpret_cast<unsigned int*>(s_x);
#pragma unroll
            for (int i = 0; i < NX; ++i) dst[tid + i * NTHREADS] = xr[i];
        }
        __syncthreads();   // x_t + h_{t-1} visible to all waves

        // ---- issue next tile's global loads; the s_wait_loadcnt lands just
        //      before next iteration's ds_store, hiding L2 latency under the
        //      whole WMMA + activation phase ----
        if (t + 1 < TSTEPS) {
            const unsigned int* src = ysin32 + ((size_t)(t + 1) * BATCH + b0) * (KIN / 2);
#pragma unroll
            for (int i = 0; i < NX; ++i) xr[i] = src[tid + i * NTHREADS];
        }

        // ---- GEMM + cell update ----
#pragma unroll
        for (int s = 0; s < 3; ++s) {
            const int mt = mt_[s], jt = jt_[s];
            v8f acc[4];
#pragma unroll
            for (int g = 0; g < 4; ++g) {
                float bb = s_b[g * HID + jt * 16 + (lane & 15)];
#pragma unroll
                for (int v = 0; v < 8; ++v) acc[g][v] = bb;
            }
            // x_t @ W_ih^T
#pragma unroll
            for (int kt = 0; kt < KIN / 32; ++kt) {
                v16bf a = ld_frag(s_x, mt * 16, kt * 32, KIN);
#pragma unroll
                for (int g = 0; g < 4; ++g) {
                    v16bf bfr = ld_frag(s_wih, g * HID + jt * 16, kt * 32, KIN);
                    acc[g] = __builtin_amdgcn_wmma_f32_16x16x32_bf16(
                        false, a, false, bfr, (short)0, acc[g], false, false);
                }
            }
            // h_{t-1} @ W_hh^T
#pragma unroll
            for (int kt = 0; kt < HID / 32; ++kt) {
                v16bf a = ld_frag(s_h, mt * 16, kt * 32, HID);
#pragma unroll
                for (int g = 0; g < 4; ++g) {
                    v16bf bfr = ld_frag(s_whh, g * HID + jt * 16, kt * 32, HID);
                    acc[g] = __builtin_amdgcn_wmma_f32_16x16x32_bf16(
                        false, a, false, bfr, (short)0, acc[g], false, false);
                }
            }
            // LSTM cell (torch gate order i,f,g,o) — fully in registers
#pragma unroll
            for (int v = 0; v < 8; ++v) {
                float ig = fast_sigmoid(acc[0][v]);
                float fg = fast_sigmoid(acc[1][v]);
                float gg = fast_tanh(acc[2][v]);
                float og = fast_sigmoid(acc[3][v]);
                float cn = fg * creg[s][v] + ig * gg;
                creg[s][v] = cn;
                hreg[s][v] = og * fast_tanh(cn);
            }
        }
        __syncthreads();   // all waves done reading h before overwrite

        // ---- write new h into LDS (D-layout: row = v + (lane>=16 ? 8:0)) ----
#pragma unroll
        for (int s = 0; s < 3; ++s) {
            const int j = jt_[s] * 16 + (lane & 15);
#pragma unroll
            for (int v = 0; v < 8; ++v) {
                int m = mt_[s] * 16 + v + ((lane & 16) ? 8 : 0);
                s_h[m * HID + j] = (__bf16)hreg[s][v];
            }
        }
        __syncthreads();

        // ---- stream h_t to global (coalesced), also hn on last step ----
        {
            unsigned int* dst = ysout32 + ((size_t)t * BATCH + b0) * (HID / 2);
            const unsigned int* hu = reinterpret_cast<const unsigned int*>(s_h);
#pragma unroll
            for (int i = 0; i < BT * HID / 2 / NTHREADS; ++i)
                dst[tid + i * NTHREADS] = hu[tid + i * NTHREADS];
            if (hn_out_u != nullptr && t == TSTEPS - 1) {
                unsigned int* hnd = reinterpret_cast<unsigned int*>(hn_out_u) +
                                    (size_t)b0 * (HID / 2);
#pragma unroll
                for (int i = 0; i < BT * HID / 2 / NTHREADS; ++i)
                    hnd[tid + i * NTHREADS] = hu[tid + i * NTHREADS];
            }
        }
    }
}

// ---------------------------------------------------------------------------
// Head: out = fc3(relu(fc2(relu(fc1(relu(hn3+hn1))))))
// ---------------------------------------------------------------------------
__global__ __launch_bounds__(NTHREADS) void k_head(
    const unsigned short* __restrict__ hn1, const unsigned short* __restrict__ hn3,
    const float* __restrict__ w1, const float* __restrict__ b1,
    const float* __restrict__ w2, const float* __restrict__ b2,
    const float* __restrict__ w3, const float* __restrict__ b3,
    float* __restrict__ out)
{
    __shared__ float sw1[16 * HID], sb1[16], sw2[8 * 16], sb2[8], sw3[8], sb3[1];
    const int tid = threadIdx.x;
    for (int i = tid; i < 16 * HID; i += NTHREADS) sw1[i] = w1[i];
    if (tid < 16) sb1[tid] = b1[tid];
    for (int i = tid; i < 128; i += NTHREADS) sw2[i] = w2[i];
    if (tid < 8) { sb2[tid] = b2[tid]; sw3[tid] = w3[tid]; }
    if (tid == 0) sb3[0] = b3[0];
    __syncthreads();

    const int b = blockIdx.x * NTHREADS + tid;
    if (b >= BATCH) return;

    float s1[16];
#pragma unroll
    for (int o = 0; o < 16; ++o) s1[o] = sb1[o];
    for (int j = 0; j < HID; ++j) {
        float z = bf2f(hn1[b * HID + j]) + bf2f(hn3[b * HID + j]);
        z = fmaxf(z, 0.0f);
#pragma unroll
        for (int o = 0; o < 16; ++o) s1[o] = __builtin_fmaf(sw1[o * HID + j], z, s1[o]);
    }
    float s2[8];
#pragma unroll
    for (int o = 0; o < 8; ++o) s2[o] = sb2[o];
#pragma unroll
    for (int i = 0; i < 16; ++i) {
        float z = fmaxf(s1[i], 0.0f);
#pragma unroll
        for (int o = 0; o < 8; ++o) s2[o] = __builtin_fmaf(sw2[o * 16 + i], z, s2[o]);
    }
    float acc = sb3[0];
#pragma unroll
    for (int o = 0; o < 8; ++o) acc = __builtin_fmaf(sw3[o], fmaxf(s2[o], 0.0f), acc);
    out[b] = acc;
}

// ---------------------------------------------------------------------------
// Host launcher
// ---------------------------------------------------------------------------
static inline size_t align256(size_t x) { return (x + 255) & ~(size_t)255; }

extern "C" void kernel_launch(void* const* d_in, const int* in_sizes, int n_in,
                              void* d_out, int out_size, void* d_ws, size_t ws_size,
                              hipStream_t stream) {
    (void)in_sizes; (void)n_in; (void)out_size; (void)ws_size;
    const float* x       = (const float*)d_in[0];
    const float* W_ih0   = (const float*)d_in[1];
    const float* W_ihR   = (const float*)d_in[2];   // [3][384][96]
    const float* W_hh    = (const float*)d_in[3];   // [4][384][96]
    const float* b_ih    = (const float*)d_in[4];   // [4][384]
    const float* b_hh    = (const float*)d_in[5];
    const float* fc1_w   = (const float*)d_in[6];
    const float* fc1_b   = (const float*)d_in[7];
    const float* fc2_w   = (const float*)d_in[8];
    const float* fc2_b   = (const float*)d_in[9];
    const float* fc3_w   = (const float*)d_in[10];
    const float* fc3_b   = (const float*)d_in[11];
    float* out = (float*)d_out;

    // workspace layout
    char* ws = (char*)d_ws;
    size_t o = 0;
    unsigned short* xpad = (unsigned short*)(ws + o); o = align256(o + (size_t)TSTEPS * BATCH * 32 * 2);
    unsigned short* ys   = (unsigned short*)(ws + o); o = align256(o + (size_t)TSTEPS * BATCH * HID * 2);
    unsigned short* wih0 = (unsigned short*)(ws + o); o = align256(o + (size_t)G4H * 32 * 2);
    unsigned short* wihR = (unsigned short*)(ws + o); o = align256(o + (size_t)3 * G4H * HID * 2);
    unsigned short* whh  = (unsigned short*)(ws + o); o = align256(o + (size_t)4 * G4H * HID * 2);
    float*          bias = (float*)(ws + o);          o = align256(o + (size_t)4 * G4H * 4);
    unsigned short* hn1  = (unsigned short*)(ws + o); o = align256(o + (size_t)BATCH * HID * 2);
    unsigned short* hn3  = (unsigned short*)(ws + o); o = align256(o + (size_t)BATCH * HID * 2);

    // --- prep: conversions ---
    {
        int n = 3 * G4H * HID;
        k_cvt_flat<<<(n + 255) / 256, 256, 0, stream>>>(W_ihR, wihR, n);
        n = 4 * G4H * HID;
        k_cvt_flat<<<(n + 255) / 256, 256, 0, stream>>>(W_hh, whh, n);
        k_cvt_wih0<<<(G4H * 32 + 255) / 256, 256, 0, stream>>>(W_ih0, wih0);
        k_bias_add<<<(4 * G4H + 255) / 256, 256, 0, stream>>>(b_ih, b_hh, bias, 4 * G4H);
        int nx = TSTEPS * BATCH * 32;
        k_cvt_x<<<nx / 256, 256, 0, stream>>>(x, xpad);
    }

    // --- LSTM layers (ys is processed in-place for layers 1..3) ---
    const unsigned lds32 = (G4H * 32 + G4H * HID + BT * HID + BT * 32) * 2 + G4H * 4;
    const unsigned lds96 = (G4H * HID + G4H * HID + BT * HID + BT * HID) * 2 + G4H * 4;
    const int nwg = BATCH / BT;

    k_lstm_layer<32><<<nwg, NTHREADS, lds32, stream>>>(
        xpad, ys, wih0, whh + 0 * G4H * HID, bias + 0 * G4H, nullptr);
    k_lstm_layer<96><<<nwg, NTHREADS, lds96, stream>>>(
        ys, ys, wihR + 0 * G4H * HID, whh + 1 * G4H * HID, bias + 1 * G4H, hn1);
    k_lstm_layer<96><<<nwg, NTHREADS, lds96, stream>>>(
        ys, ys, wihR + 1 * G4H * HID, whh + 2 * G4H * HID, bias + 2 * G4H, nullptr);
    k_lstm_layer<96><<<nwg, NTHREADS, lds96, stream>>>(
        ys, ys, wihR + 2 * G4H * HID, whh + 3 * G4H * HID, bias + 3 * G4H, hn3);

    // --- head MLP ---
    k_head<<<BATCH / NTHREADS, NTHREADS, 0, stream>>>(
        hn1, hn3, fc1_w, fc1_b, fc2_w, fc2_b, fc3_w, fc3_b, out);
}
